// HDC_20761871909028
// MI455X (gfx1250) — matrix operand order repaired and checked by reading, not compiled, not run on hardware
//
#include <hip/hip_runtime.h>

typedef __attribute__((ext_vector_type(16))) __bf16 v16bf;
typedef __attribute__((ext_vector_type(8)))  float  v8f;

#define H_    128
#define W_    512
#define CIN   64
#define COUT  64
#define NB    8
#define HALO  160                 // >= max dilation 152, multiple of 16
#define LDSW  (W_ + 2*HALO)       // 832 columns
#define ROWB  (CIN*2)             // 128 bytes of bf16 per w-column
#define NTAP  17
#define FRAG_BYTES 1024           // 32 lanes * 16 bf16 * 2B
#define NFRAG (NTAP*4*2)          // taps * co-tiles * k-halves
#define WSA_BYTES (NFRAG*FRAG_BYTES)   // 139264

// tap t=0: fused center (offset 0); t=1..16: branch (t-1)/2, left(-d)/right(+d)
constexpr int TAPOFF[NTAP] =
    {0, -48,48, -76,76, -96,96, -111,111, -124,124, -135,135, -144,144, -152,152};

__device__ __forceinline__ unsigned short f2bf(float f) {
    union { float f; unsigned u; } x; x.f = f;
    unsigned r = x.u + 0x7FFFu + ((x.u >> 16) & 1u);   // round-to-nearest-even
    return (unsigned short)(r >> 16);
}

// ---------------------------------------------------------------------------
// Prep: fuse center taps, swizzle all 17 tap matrices into WMMA A-fragment
// layout, convert to bf16; also Bsum[co] = sum_i B[i,co].
// ---------------------------------------------------------------------------
__global__ void __launch_bounds__(256)
prep_kernel(const float* __restrict__ W, const float* __restrict__ B,
            unsigned short* __restrict__ wsA, float* __restrict__ wsBias)
{
    int g = blockIdx.x * blockDim.x + threadIdx.x;
    if (g < COUT) {
        float s = 0.f;
        for (int i = 0; i < NB; ++i) s += B[i*COUT + g];
        wsBias[g] = s;
    }
    if (g >= NFRAG*32*16) return;
    int e  = g & 15;
    int L  = (g >> 4) & 31;
    int kh = (g >> 9) & 1;
    int ct = (g >> 10) & 3;
    int t  = g >> 12;
    int hf = L >> 4;
    int co = ct*16 + (L & 15);
    int ci = kh*32 + ((e & 8) ? 16 : 0) + 8*hf + (e & 7);
    float v;
    if (t == 0) {
        v = 0.f;
        for (int i = 0; i < NB; ++i) v += W[((i*COUT + co)*CIN + ci)*3 + 1];
    } else {
        int i = (t - 1) >> 1;
        int k = ((t - 1) & 1) ? 2 : 0;   // t odd -> k=0 (-d), t even -> k=2 (+d)
        v = W[((i*COUT + co)*CIN + ci)*3 + k];
    }
    wsA[g] = f2bf(v);   // linear index == frag*512 + L*16 + e by construction
}

// ---------------------------------------------------------------------------
// Main: one block per (b,h) row. Each wave owns 4 w-tiles x ALL 4 co-tiles
// (16 accumulator tiles, 128 VGPRs): B fragments are reused across the 4
// co-tiles (4x less LDS traffic) and A fragments across the 4 w-tiles
// (256B LDS + 256B L2 per WMMA). K-halves as two passes to halve liveness.
// ---------------------------------------------------------------------------
__global__ void __launch_bounds__(256)
conv_kernel(const float* __restrict__ x,
            const unsigned char* __restrict__ wsA,
            const float* __restrict__ wsBias,
            float* __restrict__ out)
{
    extern __shared__ unsigned char smem[];
    const int tid = threadIdx.x;
    const int b = blockIdx.x >> 7;
    const int h = blockIdx.x & 127;

    // zero the halo columns
    for (int idx = tid; idx < (2*HALO)*(ROWB/4); idx += 256) {
        int col = idx >> 5;                 // 32 dwords per column
        int d   = idx & 31;
        int c   = (col < HALO) ? col : col + W_;
        *(unsigned*)(smem + c*ROWB + d*4) = 0u;
    }
    // transpose-fill: global [ci][w] fp32 -> LDS [w][ci] bf16 (ci pairs packed)
    const size_t xbase = ((size_t)b*CIN)*H_*W_ + (size_t)h*W_;
    for (int it = 0; it < 64; ++it) {
        int idx = it*256 + tid;
        int w   = idx & 511;                // consecutive lanes -> coalesced
        int ci2 = idx >> 9;                 // 0..31
        float v0 = x[xbase + (size_t)(2*ci2  )*H_*W_ + w];
        float v1 = x[xbase + (size_t)(2*ci2+1)*H_*W_ + w];
        unsigned pk = (unsigned)f2bf(v0) | ((unsigned)f2bf(v1) << 16);
        *(unsigned*)(smem + (w + HALO)*ROWB + ci2*4) = pk;
    }
    __syncthreads();

    const int lane = tid & 31;
    const int wid  = tid >> 5;       // wave owns w-tiles wid*4 .. wid*4+3
    const int hf   = lane >> 4;
    const int n    = lane & 15;

    // A-fragment base (per-lane); frag(t,ct,kh) at +t*8192 + ct*2048 + kh*1024
    const unsigned char* abase = wsA + lane*32;

    // per-w-tile LDS byte base
    int bbase[4];
#pragma unroll
    for (int j = 0; j < 4; ++j)
        bbase[j] = ((wid*4 + j)*16 + n + HALO)*ROWB + hf*16;

    // accumulators: [co-tile][w-tile], init from Bsum (vector loads)
    v8f acc[4][4];
#pragma unroll
    for (int ct = 0; ct < 4; ++ct) {
        v8f bv = *(const v8f*)(wsBias + ct*16 + 8*hf);   // co = ct*16+8*hf+r
#pragma unroll
        for (int j = 0; j < 4; ++j) acc[ct][j] = bv;
    }

#pragma unroll
    for (int kh = 0; kh < 2; ++kh) {
#pragma unroll
        for (int t = 0; t < NTAP; ++t) {
            const int boff = TAPOFF[t] * ROWB + kh*64;   // literal after unroll
            v16bf a[4];
#pragma unroll
            for (int ct = 0; ct < 4; ++ct)
                a[ct] = *(const v16bf*)(abase + t*(8*FRAG_BYTES)
                                        + ct*(2*FRAG_BYTES) + kh*FRAG_BYTES);
            union { uint4 q[2]; v16bf v; } bf[4];
#pragma unroll
            for (int j = 0; j < 4; ++j) {
                const unsigned char* bp = smem + (bbase[j] + boff);
                bf[j].q[0] = *(const uint4*)(bp);
                bf[j].q[1] = *(const uint4*)(bp + 32);
            }
#pragma unroll
            for (int ct = 0; ct < 4; ++ct)
#pragma unroll
                for (int j = 0; j < 4; ++j)
                    acc[ct][j] = __builtin_amdgcn_wmma_f32_16x16x32_bf16(
                        false, a[ct], false, bf[j].v, (short)0, acc[ct][j],
                        false, false);
        }
    }

    // ReLU + store
#pragma unroll
    for (int ct = 0; ct < 4; ++ct)
#pragma unroll
        for (int j = 0; j < 4; ++j) {
            const int w = (wid*4 + j)*16 + n;
#pragma unroll
            for (int r = 0; r < 8; ++r) {
                const int co = ct*16 + r + 8*hf;
                out[(((size_t)b*COUT + co)*H_ + h)*W_ + w] =
                    fmaxf(acc[ct][j][r], 0.0f);
            }
        }
}

extern "C" void kernel_launch(void* const* d_in, const int* in_sizes, int n_in,
                              void* d_out, int out_size, void* d_ws, size_t ws_size,
                              hipStream_t stream)
{
    const float* x = (const float*)d_in[0];   // (8,64,128,512)
    const float* W = (const float*)d_in[1];   // (8,64,64,1,3)
    const float* B = (const float*)d_in[2];   // (8,64)
    float* out = (float*)d_out;               // (8,64,128,512)

    unsigned short* wsA  = (unsigned short*)d_ws;
    float*          wsBs = (float*)((char*)d_ws + WSA_BYTES);

    // allow >64KB dynamic LDS (104 KB of the 320 KB WGP pool)
    static const int lds_bytes = LDSW * ROWB;
    hipFuncSetAttribute((const void*)conv_kernel,
                        hipFuncAttributeMaxDynamicSharedMemorySize, lds_bytes);

    prep_kernel<<<(NFRAG*32*16 + 255)/256, 256, 0, stream>>>(W, B, wsA, wsBs);
    conv_kernel<<<NB*H_, 256, lds_bytes, stream>>>(x, (const unsigned char*)wsA, wsBs, out);
}
